// ContrastiveLoss_43087111913825
// MI455X (gfx1250) — compile-verified, matrix operand order checked by reading, not run on hardware
//
#include <hip/hip_runtime.h>
#include <math.h>

typedef __attribute__((ext_vector_type(16))) _Float16 v16h;
typedef __attribute__((ext_vector_type(8)))  float    v8f;
typedef int v4i_ __attribute__((vector_size(16)));   // matches builtin's V4i param

#define AS1 __attribute__((address_space(1)))
#define AS3 __attribute__((address_space(3)))

#if defined(__has_builtin)
#  if __has_builtin(__builtin_amdgcn_global_load_async_to_lds_b128)
#    define USE_ASYNC_LDS 1
#  endif
#endif
#ifndef USE_ASYNC_LDS
#  define USE_ASYNC_LDS 0
#endif

namespace {
constexpr int   kN      = 8192;   // G * CROPS
constexpr int   kD      = 128;
constexpr float kInvTau = 2.0f;   // 1 / tau, tau = 0.5
constexpr float kEps    = 1e-8f;

constexpr int BM  = 128;          // i-tile rows
constexpr int BN  = 128;          // j-tile cols
constexpr int LDA = kD + 8;       // padded LDS row stride (halves) -> kills bank conflicts
constexpr int JT  = 8;            // j-tiles swept per block
constexpr int NPART = (kN / BN / JT) * 2;  // 8 chunks * 2 waveN strips = 16 partials/row
}

// ---------------------------------------------------------------------------
// 1) Row L2-normalize: write z in f32 (exact group pass) and f16 (WMMA feed).
// ---------------------------------------------------------------------------
__global__ void __launch_bounds__(128)
normalize_kernel(const float* __restrict__ f,
                 float* __restrict__ zf, _Float16* __restrict__ zh) {
    const int row = blockIdx.x;
    const int t   = threadIdx.x;              // 128 threads = 4 waves
    const float x = f[row * kD + t];
    float s = x * x;
    #pragma unroll
    for (int o = 16; o > 0; o >>= 1) s += __shfl_xor(s, o, 32);
    __shared__ float red[4];
    if ((t & 31) == 0) red[t >> 5] = s;
    __syncthreads();
    const float nrm = sqrtf(red[0] + red[1] + red[2] + red[3]);
    const float z   = x / fmaxf(nrm, kEps);
    zf[row * kD + t] = z;
    zh[row * kD + t] = (_Float16)z;
}

// ---------------------------------------------------------------------------
// Stage one 128x128-half B tile into (padded) LDS.
// Async path: one global_load_async_to_lds_b128 per lane-iteration (8 per
// wave -> ASYNCcnt +8), completion enforced with s_wait_asynccnt + barrier.
// ---------------------------------------------------------------------------
__device__ __forceinline__ void stage_tile(_Float16* dst, const _Float16* srcTile,
                                           int tid) {
    const uint4* src = (const uint4*)srcTile;
    for (int idx = tid; idx < BN * (kD / 8); idx += 256) {
        const int row = idx >> 4, seg = idx & 15;
#if USE_ASYNC_LDS
        __builtin_amdgcn_global_load_async_to_lds_b128(
            (AS1 v4i_*)(void*)(src + idx),
            (AS3 v4i_*)(void*)&dst[row * LDA + seg * 8],
            0, 0);
#else
        *(uint4*)&dst[row * LDA + seg * 8] = src[idx];
#endif
    }
}

__device__ __forceinline__ void wait_async_le8() {
#if USE_ASYNC_LDS
    asm volatile("s_wait_asynccnt 0x8" ::: "memory");
#endif
}
__device__ __forceinline__ void wait_async_zero() {
#if USE_ASYNC_LDS
    asm volatile("s_wait_asynccnt 0x0" ::: "memory");
#endif
}

// ---------------------------------------------------------------------------
// 2) Gram row-sums of exp(z_i . z_j / tau) via f16 WMMA, flash-style:
//    never materialize the 8192x8192 matrix. Grid (64 i-tiles, 8 j-chunks).
//    A fragments live in registers for the whole block; B double-buffered
//    in LDS with async global->LDS copies overlapping the WMMA sweep.
// ---------------------------------------------------------------------------
__global__ void __launch_bounds__(256)
gram_rowsum_kernel(const _Float16* __restrict__ zh, float* __restrict__ part) {
    __shared__ _Float16 As[BM * LDA];        // ~34 KB
    __shared__ _Float16 Bs[2][BN * LDA];     // ~68 KB double buffer

    const int tid   = threadIdx.x;
    const int lane  = tid & 31;
    const int wave  = tid >> 5;          // 0..7
    const int waveM = wave & 3;          // 4 strips of 32 rows
    const int waveN = wave >> 2;         // 2 strips of 64 cols
    const int iTile = blockIdx.x;
    const int chunk = blockIdx.y;

    // Stage A tile once (plain stores; it is read many times).
    {
        const uint4* src = (const uint4*)(zh + (size_t)iTile * BM * kD);
        for (int idx = tid; idx < BM * (kD / 8); idx += 256) {
            const int row = idx >> 4, seg = idx & 15;
            *(uint4*)&As[row * LDA + seg * 8] = src[idx];
        }
    }

    // Kick off async staging of the first B tile while A settles.
    stage_tile(Bs[0], zh + (size_t)(chunk * JT) * BN * kD, tid);

    __syncthreads();   // A tile visible to all waves (dscnt drained by barrier)

    const int rRow = lane & 15;          // fragment row/col within 16
    const int kHi  = (lane >> 4) << 3;   // 0 or 8: K-half / row-half select

    // Hoist all A fragments into registers: 4 K-steps x 2 M-frags x 8 VGPRs.
    v16h aAll[4][2];
    #pragma unroll
    for (int kk = 0; kk < 4; ++kk)
        #pragma unroll
        for (int mi = 0; mi < 2; ++mi) {
            const _Float16* p = &As[(waveM * 32 + mi * 16 + rRow) * LDA + kk * 32 + kHi];
            #pragma unroll
            for (int q = 0; q < 8; ++q) { aAll[kk][mi][q] = p[q]; aAll[kk][mi][8 + q] = p[16 + q]; }
        }

    float acc[2][8];
    #pragma unroll
    for (int mi = 0; mi < 2; ++mi)
        #pragma unroll
        for (int r = 0; r < 8; ++r) acc[mi][r] = 0.0f;

    for (int t = 0; t < JT; ++t) {
        // Prefetch next tile into the other buffer (its last readers finished
        // at the end-of-iteration barrier of t-1).
        if (t + 1 < JT)
            stage_tile(Bs[(t + 1) & 1],
                       zh + (size_t)(chunk * JT + t + 1) * BN * kD, tid);

        // Async loads retire in order: <=8 outstanding => current tile done.
        if (t + 1 < JT) wait_async_le8(); else wait_async_zero();
        __syncthreads();                 // all waves' current-tile data in LDS

        const _Float16* Bcur = Bs[t & 1];

        v8f c[2][4];
        const v8f vzero = {0.f, 0.f, 0.f, 0.f, 0.f, 0.f, 0.f, 0.f};
        #pragma unroll
        for (int mi = 0; mi < 2; ++mi)
            #pragma unroll
            for (int ni = 0; ni < 4; ++ni) c[mi][ni] = vzero;

        #pragma unroll
        for (int kk = 0; kk < 4; ++kk) {
            // B fragment mirrors A (Gram product: B = Z_j^T, lane = output col)
            v16h b[4];
            #pragma unroll
            for (int ni = 0; ni < 4; ++ni) {
                const _Float16* p = &Bcur[(waveN * 64 + ni * 16 + rRow) * LDA + kk * 32 + kHi];
                #pragma unroll
                for (int q = 0; q < 8; ++q) { b[ni][q] = p[q]; b[ni][8 + q] = p[16 + q]; }
            }
            #pragma unroll
            for (int mi = 0; mi < 2; ++mi)
                #pragma unroll
                for (int ni = 0; ni < 4; ++ni)
                    c[mi][ni] = __builtin_amdgcn_wmma_f32_16x16x32_f16(
                        false, aAll[kk][mi], false, b[ni], (short)0, c[mi][ni], false, false);
        }

        // exp(S/tau) and per-row accumulate (v_exp is TRANS: co-issues with WMMA)
        #pragma unroll
        for (int mi = 0; mi < 2; ++mi)
            #pragma unroll
            for (int r = 0; r < 8; ++r) {
                float s = 0.0f;
                #pragma unroll
                for (int ni = 0; ni < 4; ++ni) s += __expf(c[mi][ni][r] * kInvTau);
                acc[mi][r] += s;
            }

        __syncthreads();   // readers done before buffer is overwritten at t+1
    }

    // Reduce the 16 column-lanes of each half, write deterministic partials.
    #pragma unroll
    for (int mi = 0; mi < 2; ++mi)
        #pragma unroll
        for (int r = 0; r < 8; ++r) {
            float s = acc[mi][r];
            s += __shfl_xor(s, 1, 32);
            s += __shfl_xor(s, 2, 32);
            s += __shfl_xor(s, 4, 32);
            s += __shfl_xor(s, 8, 32);
            if ((lane & 15) == 0) {
                const int row = iTile * BM + waveM * 32 + mi * 16 + r + kHi;
                part[(chunk * 2 + waveN) * kN + row] = s;
            }
        }
}

// ---------------------------------------------------------------------------
// 3) Per-row loss: total from partials (fixed order), exact f32 group pass.
// ---------------------------------------------------------------------------
__global__ void __launch_bounds__(256)
finalize_kernel(const float* __restrict__ zf, const float* __restrict__ part,
                float* __restrict__ loss) {
    const int i = blockIdx.x * blockDim.x + threadIdx.x;
    if (i >= kN) return;
    float total = 0.0f;
    #pragma unroll
    for (int p = 0; p < NPART; ++p) total += part[p * kN + i];

    const int base = i & ~3;            // group of 4 consecutive crops
    float grp = 0.0f;
    #pragma unroll
    for (int j = 0; j < 4; ++j) {
        float dot = 0.0f;
        for (int k = 0; k < kD; ++k) dot += zf[i * kD + k] * zf[(base + j) * kD + k];
        grp += __expf(dot * kInvTau);
    }
    const float pos = grp - __expf(kInvTau);   // subtract exp(1/tau) self term
    const float neg = total - grp;
    loss[i] = -__logf(pos / neg);
}

// ---------------------------------------------------------------------------
// 4) Deterministic single-block mean reduction (no fp atomics anywhere).
// ---------------------------------------------------------------------------
__global__ void __launch_bounds__(256)
reduce_kernel(const float* __restrict__ loss, float* __restrict__ out) {
    __shared__ float red[256];
    float s = 0.0f;
    for (int i = threadIdx.x; i < kN; i += 256) s += loss[i];
    red[threadIdx.x] = s;
    __syncthreads();
    #pragma unroll
    for (int o = 128; o > 0; o >>= 1) {
        if (threadIdx.x < o) red[threadIdx.x] += red[threadIdx.x + o];
        __syncthreads();
    }
    if (threadIdx.x == 0) out[0] = red[0] / (float)kN;
}

// ---------------------------------------------------------------------------
extern "C" void kernel_launch(void* const* d_in, const int* in_sizes, int n_in,
                              void* d_out, int out_size, void* d_ws, size_t ws_size,
                              hipStream_t stream) {
    (void)in_sizes; (void)n_in; (void)out_size; (void)ws_size;
    const float* features = (const float*)d_in[0];
    // d_in[1] (num_crops) is constant CROPS=4 with contiguous groups per setup.

    char* ws = (char*)d_ws;
    constexpr size_t ZH_OFF   = 0;
    constexpr size_t ZH_SZ    = (size_t)kN * kD * sizeof(_Float16);  // 2 MiB
    constexpr size_t ZF_OFF   = ZH_OFF + ZH_SZ;
    constexpr size_t ZF_SZ    = (size_t)kN * kD * sizeof(float);     // 4 MiB
    constexpr size_t PART_OFF = ZF_OFF + ZF_SZ;
    constexpr size_t PART_SZ  = (size_t)NPART * kN * sizeof(float);  // 512 KiB
    constexpr size_t LOSS_OFF = PART_OFF + PART_SZ;                  // 32 KiB

    _Float16* zh   = (_Float16*)(ws + ZH_OFF);
    float*    zf   = (float*)(ws + ZF_OFF);
    float*    part = (float*)(ws + PART_OFF);
    float*    loss = (float*)(ws + LOSS_OFF);

    normalize_kernel<<<kN, 128, 0, stream>>>(features, zf, zh);
    gram_rowsum_kernel<<<dim3(kN / BM, kN / (BN * JT)), 256, 0, stream>>>(zh, part);
    finalize_kernel<<<kN / 256, 256, 0, stream>>>(zf, part, loss);
    reduce_kernel<<<1, 256, 0, stream>>>(loss, (float*)d_out);
}
